// Next_Node_Probability_Calculator_for_group_28097676051158
// MI455X (gfx1250) — compile-verified, matrix operand order checked by reading, not run on hardware
//
#include <hip/hip_runtime.h>
#include <hip/hip_bf16.h>

#define B_SZ    2048
#define N_NODES 100
#define G_SZ    100
#define E_DIM   128
#define H_NUM   8
#define NPAD    112      // 7 tiles of 16 (covers N=G=100)
#define STRH    136      // f16 row stride (272B: 16B-aligned, bank-rotating)
#define STRS    116      // f32 score row stride

typedef __attribute__((ext_vector_type(16))) _Float16 v16h;
typedef __attribute__((ext_vector_type(8)))  _Float16 v8h;
typedef __attribute__((ext_vector_type(8)))  float    v8f;

// ---------------- LDS layout (bytes) ----------------
#define SZ_HALFBUF (NPAD*STRH*2)            // 30464
#define OFF_NODES  0
#define OFF_Q      (OFF_NODES + SZ_HALFBUF)
#define OFF_K      (OFF_Q     + SZ_HALFBUF)
#define OFF_OUTC   (OFF_K     + SZ_HALFBUF) // last-node embeds, reused as out_concat
#define OFF_VT     (OFF_OUTC  + SZ_HALFBUF)
#define SZ_VT      (128*STRH*2)             // 34816 (V transposed: [channel][node])
#define OFF_WTS    (OFF_VT    + SZ_VT)      // softmax weights f16 [g][n], 30464
#define OFF_SC     (OFF_WTS   + SZ_HALFBUF)
#define SZ_SC      (NPAD*STRS*4)            // 51968 (raw scores f32)
#define OFF_RI     (OFF_SC    + SZ_SC)      // f32[128]: graph mean, later 1/rowsum
#define OFF_GQ     (OFF_RI    + 512)        // f32[128]: q_graph
#define LDS_BYTES  (OFF_GQ    + 512)        // 240128 <= 320KB WGP LDS

__device__ __forceinline__ v8h ld8(const _Float16* p) { return *(const v8h*)p; }

__device__ __forceinline__ v16h cat16(v8h lo, v8h hi) {
  v16h r;
#pragma unroll
  for (int i = 0; i < 8; ++i) { r[i] = lo[i]; r[i + 8] = hi[i]; }
  return r;
}
__device__ __forceinline__ v16h cat16z(v8h lo) {
  v16h r;
#pragma unroll
  for (int i = 0; i < 8; ++i) { r[i] = lo[i]; r[i + 8] = (_Float16)0.f; }
  return r;
}
#define WMMA_F16(a, b, c) \
  __builtin_amdgcn_wmma_f32_16x16x32_f16(false, (a), false, (b), (short)0, (c), false, false)

// Pre-transpose weights to f16 [out][in] so WMMA B fragments are contiguous.
// Also folds Wq_first + Wq_last into one matrix (q uses the same input for both).
__global__ __launch_bounds__(256) void prep_weights(
    const float* __restrict__ Wq_first, const float* __restrict__ Wq_last,
    const float* __restrict__ Wk, const float* __restrict__ Wv,
    const float* __restrict__ Wc, _Float16* __restrict__ ws) {
  int idx = blockIdx.x * 256 + threadIdx.x;     // 0..16383
  int n = idx >> 7, k = idx & 127;
  int src = k * 128 + n;                        // W is [in,out] -> transpose
  ws[idx]         = (_Float16)(Wq_first[src] + Wq_last[src]);
  ws[idx + 16384] = (_Float16)Wk[src];
  ws[idx + 32768] = (_Float16)Wv[src];
  ws[idx + 49152] = (_Float16)Wc[src];
}

__global__ __launch_bounds__(256) void pomo_decode_step(
    const float* __restrict__ enc_nodes,   // [B,N,E]
    const float* __restrict__ enc_last,    // [B,G,E]
    const float* __restrict__ ninf_mask,   // [B,G,N]
    const float* __restrict__ Wq_graph,    // [E,E]
    const float* __restrict__ b_comb,      // [E]
    const _Float16* __restrict__ w16,      // Wqfl_t | Wk_t | Wv_t | Wc_t
    float* __restrict__ out)               // [B,G,N]
{
  extern __shared__ char smem[];
  _Float16* sN  = (_Float16*)(smem + OFF_NODES);
  _Float16* sQ  = (_Float16*)(smem + OFF_Q);
  _Float16* sK  = (_Float16*)(smem + OFF_K);
  _Float16* sO  = (_Float16*)(smem + OFF_OUTC);
  _Float16* sVT = (_Float16*)(smem + OFF_VT);
  _Float16* sW  = (_Float16*)(smem + OFF_WTS);
  float*    sS  = (float*)(smem + OFF_SC);
  float*    sR  = (float*)(smem + OFF_RI);
  float*    sGQ = (float*)(smem + OFF_GQ);

  const int b    = blockIdx.x;
  const int t    = threadIdx.x;
  const int lane = t & 31;
  const int wave = t >> 5;           // 8 waves (wave32)
  const int lr   = lane & 15;
  const int hi8  = (lane >> 4) * 8;  // K sub-offset AND C-row sub-offset

  const float* en = enc_nodes + (size_t)b * N_NODES * E_DIM;
  const float* el = enc_last  + (size_t)b * G_SZ   * E_DIM;
  const _Float16* Wqfl_t = w16;
  const _Float16* Wk_t   = w16 + 16384;
  const _Float16* Wv_t   = w16 + 32768;
  const _Float16* Wc_t   = w16 + 49152;

  // ---- stage inputs to LDS as f16, zero-padded to 112 rows ----
  for (int idx = t; idx < NPAD * E_DIM; idx += 256) {
    int r = idx >> 7, c = idx & 127;
    sN[r * STRH + c] = (_Float16)((r < N_NODES) ? en[r * E_DIM + c] : 0.f);
    sO[r * STRH + c] = (_Float16)((r < G_SZ)    ? el[r * E_DIM + c] : 0.f);
  }
  for (int idx = t; idx < 128 * 24; idx += 256) {   // zero V^T node-pad 112..135
    int ch = idx / 24, nd = idx - ch * 24;
    sVT[ch * STRH + 112 + nd] = (_Float16)0.f;
  }
  // graph mean + q_graph (f32, small)
  if (t < E_DIM) {
    float s = 0.f;
    for (int r = 0; r < N_NODES; ++r) s += en[r * E_DIM + t];
    sR[t] = s * (1.0f / N_NODES);
  }
  __syncthreads();
  if (t < E_DIM) {
    float s = 0.f;
    for (int e = 0; e < E_DIM; ++e) s += sR[e] * Wq_graph[e * E_DIM + t];
    sGQ[t] = s;
  }
  __syncthreads();

  // ---- projections: each wave owns 16 output channels ----
  {
    const int col = wave * 16 + lr;
    { // K = nodes @ Wk
      v8f acc[7] = {};
#pragma unroll
      for (int kc = 0; kc < 4; ++kc) {
        const _Float16* wp = Wk_t + col * 128 + kc * 32 + hi8;
        v16h bf = cat16(ld8(wp), ld8(wp + 16));
#pragma unroll
        for (int mt = 0; mt < 7; ++mt) {
          const _Float16* ap = sN + (mt * 16 + lr) * STRH + kc * 32 + hi8;
          acc[mt] = WMMA_F16(cat16(ld8(ap), ld8(ap + 16)), bf, acc[mt]);
        }
      }
#pragma unroll
      for (int mt = 0; mt < 7; ++mt)
#pragma unroll
        for (int r = 0; r < 8; ++r)
          sK[(mt * 16 + r + hi8) * STRH + col] = (_Float16)acc[mt][r];
    }
    { // V^T = (nodes @ Wv)^T  (store channel-major for the PV GEMM)
      v8f acc[7] = {};
#pragma unroll
      for (int kc = 0; kc < 4; ++kc) {
        const _Float16* wp = Wv_t + col * 128 + kc * 32 + hi8;
        v16h bf = cat16(ld8(wp), ld8(wp + 16));
#pragma unroll
        for (int mt = 0; mt < 7; ++mt) {
          const _Float16* ap = sN + (mt * 16 + lr) * STRH + kc * 32 + hi8;
          acc[mt] = WMMA_F16(cat16(ld8(ap), ld8(ap + 16)), bf, acc[mt]);
        }
      }
#pragma unroll
      for (int mt = 0; mt < 7; ++mt)
#pragma unroll
        for (int r = 0; r < 8; ++r)
          sVT[col * STRH + mt * 16 + r + hi8] = (_Float16)acc[mt][r];
    }
    { // Q = q_graph + last @ (Wq_first + Wq_last)
      v8f acc[7] = {};
#pragma unroll
      for (int kc = 0; kc < 4; ++kc) {
        const _Float16* wp = Wqfl_t + col * 128 + kc * 32 + hi8;
        v16h bf = cat16(ld8(wp), ld8(wp + 16));
#pragma unroll
        for (int mt = 0; mt < 7; ++mt) {
          const _Float16* ap = sO + (mt * 16 + lr) * STRH + kc * 32 + hi8;
          acc[mt] = WMMA_F16(cat16(ld8(ap), ld8(ap + 16)), bf, acc[mt]);
        }
      }
      const float gqc = sGQ[col];
#pragma unroll
      for (int mt = 0; mt < 7; ++mt)
#pragma unroll
        for (int r = 0; r < 8; ++r)
          sQ[(mt * 16 + r + hi8) * STRH + col] = (_Float16)(acc[mt][r] + gqc);
    }
  }
  __syncthreads();

  // ---- multi-head attention (D=16, K padded to 32 with zeros) ----
  for (int h = 0; h < H_NUM; ++h) {
    const int cb = h * 16;
    for (int p = wave; p < 49; p += 8) {          // QK^T tiles
      const int mt = p / 7, nt = p - mt * 7;
      v16h af = cat16z(ld8(sQ + (mt * 16 + lr) * STRH + cb + hi8));
      v16h bf = cat16z(ld8(sK + (nt * 16 + lr) * STRH + cb + hi8));
      v8f c = {};
      c = WMMA_F16(af, bf, c);
#pragma unroll
      for (int r = 0; r < 8; ++r)
        sS[(mt * 16 + r + hi8) * STRS + nt * 16 + lr] = c[r];
    }
    __syncthreads();
    if (t < NPAD) {                               // row softmax
      const int g = t;
      const float* mrow = ninf_mask + ((size_t)b * G_SZ + g) * N_NODES;
      float mx = -3.0e38f;
      for (int n = 0; n < N_NODES; ++n) {
        float s = sS[g * STRS + n] * 0.25f + ((g < G_SZ) ? mrow[n] : 0.f);
        sS[g * STRS + n] = s;
        mx = fmaxf(mx, s);
      }
      float sum = 0.f;
      for (int n = 0; n < N_NODES; ++n) {
        float e = __expf(sS[g * STRS + n] - mx);
        sum += e;
        sW[g * STRH + n] = (_Float16)e;
      }
      for (int n = N_NODES; n < 128; ++n) sW[g * STRH + n] = (_Float16)0.f;
      sR[g] = 1.0f / sum;
    }
    __syncthreads();
    if (wave < 7) {                               // out = softmax @ V (16 cols)
      const int mt = wave;
      v8f acc = {};
#pragma unroll
      for (int kc = 0; kc < 4; ++kc) {
        const _Float16* ap = sW + (mt * 16 + lr) * STRH + kc * 32 + hi8;
        const _Float16* bp = sVT + (cb + lr) * STRH + kc * 32 + hi8;
        acc = WMMA_F16(cat16(ld8(ap), ld8(ap + 16)),
                       cat16(ld8(bp), ld8(bp + 16)), acc);
      }
#pragma unroll
      for (int r = 0; r < 8; ++r) {
        int row = mt * 16 + r + hi8;
        sO[row * STRH + cb + lr] = (_Float16)(acc[r] * sR[row]);
      }
    }
    __syncthreads();
  }

  // ---- combine: mh = out_concat @ W_comb + b_comb  -> sQ ----
  {
    const int col = wave * 16 + lr;
    const float bc = b_comb[col];
    v8f acc[7] = {};
#pragma unroll
    for (int kc = 0; kc < 4; ++kc) {
      const _Float16* wp = Wc_t + col * 128 + kc * 32 + hi8;
      v16h bf = cat16(ld8(wp), ld8(wp + 16));
#pragma unroll
      for (int mt = 0; mt < 7; ++mt) {
        const _Float16* ap = sO + (mt * 16 + lr) * STRH + kc * 32 + hi8;
        acc[mt] = WMMA_F16(cat16(ld8(ap), ld8(ap + 16)), bf, acc[mt]);
      }
    }
#pragma unroll
    for (int mt = 0; mt < 7; ++mt)
#pragma unroll
      for (int r = 0; r < 8; ++r)
        sQ[(mt * 16 + r + hi8) * STRH + col] = (_Float16)(acc[mt][r] + bc);
  }
  __syncthreads();

  // ---- pointer scores: mh @ nodes^T ----
  for (int p = wave; p < 49; p += 8) {
    const int mt = p / 7, nt = p - mt * 7;
    v8f acc = {};
#pragma unroll
    for (int kc = 0; kc < 4; ++kc) {
      const _Float16* ap = sQ + (mt * 16 + lr) * STRH + kc * 32 + hi8;
      const _Float16* bp = sN + (nt * 16 + lr) * STRH + kc * 32 + hi8;
      acc = WMMA_F16(cat16(ld8(ap), ld8(ap + 16)),
                     cat16(ld8(bp), ld8(bp + 16)), acc);
    }
#pragma unroll
    for (int r = 0; r < 8; ++r)
      sS[(mt * 16 + r + hi8) * STRS + nt * 16 + lr] = acc[r];
  }
  __syncthreads();

  // ---- tanh clip + mask + softmax ----
  if (t < G_SZ) {
    const int g = t;
    const float* mrow = ninf_mask + ((size_t)b * G_SZ + g) * N_NODES;
    float mx = -3.0e38f;
    for (int n = 0; n < N_NODES; ++n) {
      float s = 10.0f * tanhf(sS[g * STRS + n] * 0.08838834764831845f) + mrow[n];
      sS[g * STRS + n] = s;
      mx = fmaxf(mx, s);
    }
    float sum = 0.f;
    for (int n = 0; n < N_NODES; ++n) {
      float e = __expf(sS[g * STRS + n] - mx);
      sS[g * STRS + n] = e;
      sum += e;
    }
    sR[g] = 1.0f / sum;
  }
  __syncthreads();
  float* ob = out + (size_t)b * G_SZ * N_NODES;
  for (int idx = t; idx < G_SZ * N_NODES; idx += 256) {
    int g = idx / N_NODES, n = idx - g * N_NODES;
    ob[idx] = sS[g * STRS + n] * sR[g];
  }
}

extern "C" void kernel_launch(void* const* d_in, const int* in_sizes, int n_in,
                              void* d_out, int out_size, void* d_ws, size_t ws_size,
                              hipStream_t stream) {
  (void)in_sizes; (void)n_in; (void)out_size; (void)ws_size;
  const float* enc_nodes = (const float*)d_in[0];
  const float* enc_last  = (const float*)d_in[1];
  const float* mask      = (const float*)d_in[2];
  const float* Wq_graph  = (const float*)d_in[3];
  const float* Wq_first  = (const float*)d_in[4];
  const float* Wq_last   = (const float*)d_in[5];
  const float* Wk        = (const float*)d_in[6];
  const float* Wv        = (const float*)d_in[7];
  const float* Wc        = (const float*)d_in[8];
  const float* b_comb    = (const float*)d_in[9];
  _Float16* w16 = (_Float16*)d_ws;   // 4 * 128*128 f16 = 128 KB

  prep_weights<<<64, 256, 0, stream>>>(Wq_first, Wq_last, Wk, Wv, Wc, w16);
  pomo_decode_step<<<B_SZ, 256, LDS_BYTES, stream>>>(
      enc_nodes, enc_last, mask, Wq_graph, b_comb, w16, (float*)d_out);
}
